// PytorchTemporalMemory_72155450573341
// MI455X (gfx1250) — matrix operand order repaired and checked by reading, not compiled, not run on hardware
//
#include <hip/hip_runtime.h>

typedef __attribute__((ext_vector_type(16))) _Float16 v16h;
typedef __attribute__((ext_vector_type(8)))  float    v8f;

#define COLUMN_COUNT 1024
#define CELLS_PER_COLUMN 8
#define NUM_CELLS 8192
#define KSPLIT 4
#define KCHUNK 32
#define KRANGE (NUM_CELLS / KSPLIT)      // 2048
#define CHUNKS (KRANGE / KCHUNK)         // 64

// workspace layout (bytes):
//   adj      float[8192]      @ 0       (32 KB)   per-column permanence delta
//   actH     _Float16[8192]   @ 32768   (16 KB)   new_active as f16 (B operand)
//   partial  float[4][8192]   @ 49152   (128 KB)  K-split partial presyn sums

// ---------------------------------------------------------------------------
// Kernel A: new_active, anomaly, adj[], actH[], zero partials. One block.
// ---------------------------------------------------------------------------
__global__ void __launch_bounds__(256)
tm_prepare(const float* __restrict__ active_columns,
           const float* __restrict__ predictive_cells,
           float* __restrict__ out_new_active,
           float* __restrict__ out_anomaly,
           float* __restrict__ adj,
           _Float16* __restrict__ actH,
           float* __restrict__ partial) {
  __shared__ float s_na[256];
  __shared__ float s_np[256];
  const int t = threadIdx.x;
  float na_cnt = 0.0f, np_cnt = 0.0f;

  for (int c = t; c < COLUMN_COUNT; c += 256) {
    const float act = active_columns[c];
    const bool active = act > 0.0f;
    float pv[CELLS_PER_COLUMN];
    float ps = 0.0f;
#pragma unroll
    for (int k = 0; k < CELLS_PER_COLUMN; ++k) {
      pv[k] = predictive_cells[c * CELLS_PER_COLUMN + k];
      ps += pv[k];
    }
    const bool has_pred = ps > 0.0f;
    if (active) {
      na_cnt += 1.0f;
      if (has_pred) np_cnt += 1.0f;
    }
#pragma unroll
    for (int k = 0; k < CELLS_PER_COLUMN; ++k) {
      const float na = active ? (has_pred ? pv[k] : 1.0f) : 0.0f;
      const int idx = c * CELLS_PER_COLUMN + k;
      out_new_active[idx] = na;
      actH[idx] = (_Float16)na;
      // delta - decay per column: na binary -> 0.1*na - 0.01*(1-na)
      adj[idx] = (na > 0.0f) ? 0.1f : -0.01f;
    }
  }

  // zero K-split partial accumulators
  for (int i = t; i < KSPLIT * NUM_CELLS; i += 256) partial[i] = 0.0f;

  s_na[t] = na_cnt;
  s_np[t] = np_cnt;
  __syncthreads();
  for (int s = 128; s > 0; s >>= 1) {
    if (t < s) { s_na[t] += s_na[t + s]; s_np[t] += s_np[t + s]; }
    __syncthreads();
  }
  if (t == 0) {
    out_anomaly[0] = 1.0f - s_np[0] / fmaxf(s_na[0], 1.0f);
  }
}

// ---------------------------------------------------------------------------
// Kernel B: fused (connected @ new_active) via v_wmma_f32_16x16x32_f16 and
// branch-free permanence update, single streaming pass over W.
// Wave w: rows [16*(w/KSPLIT) .. +15], K range [2048*(w%KSPLIT) .. +2047].
// A operand (16x32 f16): lane m<16 holds row m, K {0..7,16..23};
//                        lane m+16 holds row m, K {8..15,24..31}.
// B operand (32x16 f16): new_active replicated across N; lanes 0-15 hold
//                        K {0..7,16..23}, lanes 16-31 hold K {8..15,24..31}.
// adj/act staged in LDS once per block; update = clamp(fma(adj, prev, w)),
// exact because prev is binary and clamp is identity on [0,1].
// ---------------------------------------------------------------------------
__global__ void __launch_bounds__(256)
tm_matvec_learn(const float* __restrict__ W,
                const float* __restrict__ prev_active,
                const float* __restrict__ adj,
                const _Float16* __restrict__ actH,
                float* __restrict__ Wout,
                float* __restrict__ partial) {
  __shared__ float    s_adj[NUM_CELLS];   // 32 KB
  __shared__ _Float16 s_act[NUM_CELLS];   // 16 KB

  // stage adj + act into LDS (vectorized b128 copies)
  {
    float4*       sa4 = reinterpret_cast<float4*>(s_adj);
    const float4* ga4 = reinterpret_cast<const float4*>(adj);
    for (int i = threadIdx.x; i < NUM_CELLS / 4; i += 256) sa4[i] = ga4[i];
    float4*       sh4 = reinterpret_cast<float4*>(s_act);
    const float4* gh4 = reinterpret_cast<const float4*>(actH);
    for (int i = threadIdx.x; i < NUM_CELLS / 8; i += 256) sh4[i] = gh4[i];
  }
  __syncthreads();

  const int lane = threadIdx.x & 31;
  const int gwave = blockIdx.x * 8 + (threadIdx.x >> 5);   // 0..2047
  const int rowGroup = gwave >> 2;                          // 0..511
  const int ks = gwave & (KSPLIT - 1);                      // 0..3
  const int rowBase = rowGroup * 16;
  const int kstart = ks * KRANGE;
  const bool hi = lane >= 16;
  const int row = rowBase + (lane & 15);

  const float prev = prev_active[row];                      // binary 0/1

  const float* __restrict__ wrow = W + (size_t)row * NUM_CELLS;
  float* __restrict__ worow = Wout + (size_t)row * NUM_CELLS;

  v8f c;
#pragma unroll
  for (int r = 0; r < 8; ++r) c[r] = 0.0f;

  for (int it = 0; it < CHUNKS; ++it) {
    const int kb = kstart + it * KCHUNK;
    const int off0 = kb + (hi ? 8 : 0);
    const int off1 = off0 + 16;

    const float4 w00 = *reinterpret_cast<const float4*>(wrow + off0);
    const float4 w01 = *reinterpret_cast<const float4*>(wrow + off0 + 4);
    const float4 w10 = *reinterpret_cast<const float4*>(wrow + off1);
    const float4 w11 = *reinterpret_cast<const float4*>(wrow + off1 + 4);

    if (it + 1 < CHUNKS)
      __builtin_prefetch(wrow + off0 + KCHUNK, 0, 1);  // global_prefetch_b8

    const float4 a00 = *reinterpret_cast<const float4*>(s_adj + off0);
    const float4 a01 = *reinterpret_cast<const float4*>(s_adj + off0 + 4);
    const float4 a10 = *reinterpret_cast<const float4*>(s_adj + off1);
    const float4 a11 = *reinterpret_cast<const float4*>(s_adj + off1 + 4);

    float w0[8] = {w00.x, w00.y, w00.z, w00.w, w01.x, w01.y, w01.z, w01.w};
    float w1[8] = {w10.x, w10.y, w10.z, w10.w, w11.x, w11.y, w11.z, w11.w};
    float d0[8] = {a00.x, a00.y, a00.z, a00.w, a01.x, a01.y, a01.z, a01.w};
    float d1[8] = {a10.x, a10.y, a10.z, a10.w, a11.x, a11.y, a11.z, a11.w};

    // A operand: connected = (W >= 0.5) as f16 0/1
    v16h a;
#pragma unroll
    for (int i = 0; i < 8; ++i) {
      a[i]     = (w0[i] >= 0.5f) ? (_Float16)1.0f : (_Float16)0.0f;
      a[8 + i] = (w1[i] >= 0.5f) ? (_Float16)1.0f : (_Float16)0.0f;
    }

    // B operand: f16 activities from LDS, K {8h..8h+7, 16+8h..16+8h+7} (matches A)
    const int bo = kb + (hi ? 8 : 0);
    v16h b;
#pragma unroll
    for (int i = 0; i < 8; ++i) {
      b[i]     = s_act[bo + i];
      b[8 + i] = s_act[bo + 16 + i];
    }

    c = __builtin_amdgcn_wmma_f32_16x16x32_f16(
        /*neg_a=*/false, a, /*neg_b=*/false, b,
        /*c_mod=*/(short)0, c, /*reuse_a=*/false, /*reuse_b=*/false);

    // branch-free learn update: new = clamp01(w + adj*prev)
#pragma unroll
    for (int i = 0; i < 8; ++i) {
      const float u0 = fminf(fmaxf(fmaf(d0[i], prev, w0[i]), 0.0f), 1.0f);
      const float u1 = fminf(fmaxf(fmaf(d1[i], prev, w1[i]), 0.0f), 1.0f);
      worow[off0 + i] = u0;
      worow[off1 + i] = u1;
    }
  }

  // D layout: VGPR r, lanes 0-15 -> M=r; lanes 16-31 -> M=8+r. Columns are
  // identical (B replicated), so lanes 0 and 16 publish the 16 row sums.
  if ((lane & 15) == 0) {
    const int mb = hi ? 8 : 0;
#pragma unroll
    for (int r = 0; r < 8; ++r)
      partial[ks * NUM_CELLS + rowBase + mb + r] = c[r];
  }
}

// ---------------------------------------------------------------------------
// Kernel C: reduce K-split partials, apply activation threshold.
// ---------------------------------------------------------------------------
__global__ void __launch_bounds__(256)
tm_predict(const float* __restrict__ partial, float* __restrict__ out_pred) {
  const int i = blockIdx.x * 256 + threadIdx.x;
  if (i < NUM_CELLS) {
    const float s = partial[i] + partial[NUM_CELLS + i] +
                    partial[2 * NUM_CELLS + i] + partial[3 * NUM_CELLS + i];
    out_pred[i] = (s >= 13.0f) ? 1.0f : 0.0f;
  }
}

extern "C" void kernel_launch(void* const* d_in, const int* in_sizes, int n_in,
                              void* d_out, int out_size, void* d_ws, size_t ws_size,
                              hipStream_t stream) {
  const float* active_columns   = (const float*)d_in[0];  // [1024]
  const float* segment_weights  = (const float*)d_in[1];  // [8192*8192]
  const float* predictive_cells = (const float*)d_in[2];  // [8192]
  const float* prev_active      = (const float*)d_in[3];  // [8192]

  float* out = (float*)d_out;
  float* out_new_active = out;                      // [8192]
  float* out_new_pred   = out + NUM_CELLS;          // [8192]
  float* out_anomaly    = out + 2 * NUM_CELLS;      // [1]
  float* out_weights    = out + 2 * NUM_CELLS + 1;  // [8192*8192]

  char* ws = (char*)d_ws;
  float*     adj     = (float*)(ws);
  _Float16*  actH    = (_Float16*)(ws + 32768);
  float*     partial = (float*)(ws + 49152);

  tm_prepare<<<1, 256, 0, stream>>>(active_columns, predictive_cells,
                                    out_new_active, out_anomaly,
                                    adj, actH, partial);

  // 2048 waves = 256 blocks x 8 waves; each wave: 16 rows x 2048-K slice
  tm_matvec_learn<<<256, 256, 0, stream>>>(segment_weights, prev_active,
                                           adj, actH, out_weights, partial);

  tm_predict<<<NUM_CELLS / 256, 256, 0, stream>>>(partial, out_new_pred);
}
